// WanSelfAttention_77584289235616
// MI455X (gfx1250) — compile-verified, matrix operand order checked by reading, not run on hardware
//
#include <hip/hip_runtime.h>
#include <hip/hip_bf16.h>
#include <math.h>

// Problem constants (B=1)
#define SEQ 4096
#define DIM 1536
#define NH  12
#define HD  128

typedef __attribute__((ext_vector_type(16))) __bf16 v16bf;
typedef __attribute__((ext_vector_type(8)))  float  v8f;
typedef __attribute__((ext_vector_type(4)))  int    v4i;
typedef __attribute__((address_space(1))) v4i gl_v4i;   // global-AS int4
typedef __attribute__((address_space(3))) v4i ld_v4i;   // LDS-AS int4

// Async global->LDS copy support (gfx1250): guarded so either toolchain compiles.
#ifndef HAVE_ASYNC_LDS
#if defined(__has_builtin)
#if __has_builtin(__builtin_amdgcn_global_load_async_to_lds_b128)
#define HAVE_ASYNC_LDS 1
#else
#define HAVE_ASYNC_LDS 0
#endif
#else
#define HAVE_ASYNC_LDS 0
#endif
#endif

__device__ __forceinline__ void wait_asynccnt0() {
#if defined(__has_builtin)
#if __has_builtin(__builtin_amdgcn_s_wait_asynccnt)
  __builtin_amdgcn_s_wait_asynccnt(0);
  return;
#else
  asm volatile("s_wait_asynccnt 0x0" ::: "memory");
  return;
#endif
#else
  asm volatile("s_wait_asynccnt 0x0" ::: "memory");
#endif
}

// ---------------------------------------------------------------------------
// Fragment helpers for V_WMMA_F32_16X16X32_BF16 (wave32).
// A (16x32): lane l holds row (l%16); element e holds
//   K(e) = (e>=8 ? 16 : 0) + 8*(l/16) + (e%8)  -> two 8-contiguous runs.
// B (32x16): lane l holds column (l%16); same K(e) ordering.
// C/D (16x16, f32): lane l holds col (l%16), rows m = r + 8*(l/16).
// ---------------------------------------------------------------------------

// A fragment from fp32 row-major (convert to bf16 on the fly).
__device__ __forceinline__ v16bf frag_a_f32(const float* __restrict__ A, int lda,
                                            int row, int k0, int half) {
  v16bf a;
  const float* p = A + (size_t)row * lda + k0 + 8 * half;
#pragma unroll
  for (int e = 0; e < 8; ++e) a[e] = (__bf16)p[e];
#pragma unroll
  for (int e = 0; e < 8; ++e) a[8 + e] = (__bf16)p[16 + e];
  return a;
}

// A/B fragment from bf16 with contiguous K runs (global or LDS pointer).
__device__ __forceinline__ v16bf frag_bf16_contig(const __bf16* p) {
  v16bf a;
#pragma unroll
  for (int e = 0; e < 8; ++e) a[e] = p[e];
#pragma unroll
  for (int e = 0; e < 8; ++e) a[8 + e] = p[16 + e];
  return a;
}

// ---------------------------------------------------------------------------
// Kernel 0: W [K,N] fp32 -> Wt [N,K] bf16  (LDS-tiled transpose, 32x32 tiles)
// ---------------------------------------------------------------------------
__global__ __launch_bounds__(256) void transpose_cvt_kernel(
    const float* __restrict__ W, __bf16* __restrict__ Wt, int K, int N) {
  __shared__ float tile[32][33];
  const int tx = threadIdx.x & 31;
  const int ty = threadIdx.x >> 5;               // 0..7
  const int k0 = blockIdx.x * 32;
  const int n0 = blockIdx.y * 32;
#pragma unroll
  for (int i = 0; i < 32; i += 8)
    tile[ty + i][tx] = W[(size_t)(k0 + ty + i) * N + n0 + tx];
  __syncthreads();
#pragma unroll
  for (int i = 0; i < 32; i += 8)
    Wt[(size_t)(n0 + ty + i) * K + k0 + tx] = (__bf16)tile[tx][ty + i];
}

// ---------------------------------------------------------------------------
// Kernel 1: C[M,N] = A[M,K](fp32) @ Wt[N,K](bf16, pre-transposed) + bias
// One wave computes a 64x64 tile: 16 accumulators, 16 WMMAs per k-step
// against 4 A-frag + 4 B-frag loads (all contiguous).
// ---------------------------------------------------------------------------
__global__ __launch_bounds__(32) void gemm64_kernel(
    const float* __restrict__ A, const __bf16* __restrict__ Wt,
    const float* __restrict__ bias, float* __restrict__ C,
    int M, int K, int N) {
  const int lane = threadIdx.x & 31;
  const int half = lane >> 4;
  const int nn   = lane & 15;
  const int m0 = blockIdx.x * 64;
  const int n0 = blockIdx.y * 64;

  v8f zero = {};
  v8f acc[4][4];
#pragma unroll
  for (int mi = 0; mi < 4; ++mi)
#pragma unroll
    for (int ni = 0; ni < 4; ++ni) acc[mi][ni] = zero;

  for (int k0 = 0; k0 < K; k0 += 32) {
    v16bf a[4];
#pragma unroll
    for (int mi = 0; mi < 4; ++mi)
      a[mi] = frag_a_f32(A, K, m0 + mi * 16 + nn, k0, half);
#pragma unroll
    for (int ni = 0; ni < 4; ++ni) {
      v16bf b = frag_bf16_contig(Wt + (size_t)(n0 + ni * 16 + nn) * K + k0 + 8 * half);
#pragma unroll
      for (int mi = 0; mi < 4; ++mi)
        acc[mi][ni] = __builtin_amdgcn_wmma_f32_16x16x32_bf16(
            false, a[mi], false, b, (short)0, acc[mi][ni], false, false);
    }
  }

#pragma unroll
  for (int ni = 0; ni < 4; ++ni) {
    const float bv = bias[n0 + ni * 16 + nn];
#pragma unroll
    for (int mi = 0; mi < 4; ++mi) {
#pragma unroll
      for (int r = 0; r < 8; ++r) {
        const int m = m0 + mi * 16 + 8 * half + r;
        C[(size_t)m * N + n0 + ni * 16 + nn] = acc[mi][ni][r] + bv;
      }
    }
  }
}

// ---------------------------------------------------------------------------
// Kernel 2: per-row RMSNorm (+gain) + RoPE, emit bf16.
//   mode==1: out[s*DIM + d] = bf16(rope(rms(y)*g))      (Q, K)
//   mode==0: out[(h*HD+d)*SEQ + s] = bf16(y)  transposed (V)
// ---------------------------------------------------------------------------
__global__ __launch_bounds__(256) void norm_rope_cvt_kernel(
    const float* __restrict__ Y, const float* __restrict__ g,
    const float* __restrict__ fcos, const float* __restrict__ fsin,
    __bf16* __restrict__ out, int mode_norm_rope) {
  const int s = blockIdx.x;
  const float* y = Y + (size_t)s * DIM;
  const int base = threadIdx.x * 6;

  float vals[6];
  float ss = 0.f;
#pragma unroll
  for (int j = 0; j < 6; ++j) { vals[j] = y[base + j]; ss += vals[j] * vals[j]; }

  __shared__ float red[256];
  red[threadIdx.x] = ss;
  __syncthreads();
  for (int off = 128; off > 0; off >>= 1) {
    if (threadIdx.x < off) red[threadIdx.x] += red[threadIdx.x + off];
    __syncthreads();
  }
  const float rms = rsqrtf(red[0] / (float)DIM + 1e-6f);

  if (mode_norm_rope) {
#pragma unroll
    for (int p = 0; p < 3; ++p) {
      const int d  = base + 2 * p;
      const float re = vals[2 * p]     * rms * g[d];
      const float im = vals[2 * p + 1] * rms * g[d + 1];
      const int i = (d % HD) >> 1;
      const float c  = fcos[s * (HD / 2) + i];
      const float sn = fsin[s * (HD / 2) + i];
      out[(size_t)s * DIM + d]     = (__bf16)(re * c - im * sn);
      out[(size_t)s * DIM + d + 1] = (__bf16)(re * sn + im * c);
    }
  } else {
#pragma unroll
    for (int j = 0; j < 6; ++j) {
      const int d  = base + j;
      const int h  = d / HD;
      const int dd = d % HD;
      out[(size_t)(h * HD + dd) * SEQ + s] = (__bf16)vals[j];
    }
  }
}

// ---------------------------------------------------------------------------
// Kernel 3: flash attention. 4 waves / block, one head, 64 queries per block
// (16 per wave). Per 32-key block the K tile (32x128) and V tile (128x32,
// from the pre-transposed vt) are staged cooperatively into LDS — via
// global_load_async_to_lds_b128 when the toolchain exposes it.
//   qb, kb: bf16 [s, NH*HD];  vt: bf16 [NH*HD, SEQ];  O: fp32 [s, NH*HD]
// ---------------------------------------------------------------------------
__global__ __launch_bounds__(128) void attn_kernel(
    const __bf16* __restrict__ qb, const __bf16* __restrict__ kb,
    const __bf16* __restrict__ vt, float* __restrict__ O) {
  const int h    = blockIdx.y;
  const int wid  = threadIdx.x >> 5;           // wave id 0..3
  const int q0   = blockIdx.x * 64 + wid * 16;
  const int lane = threadIdx.x & 31;
  const int half = lane >> 4;
  const int nn   = lane & 15;
  const int tid  = threadIdx.x;
  const float scale = 0.08838834764831845f;    // 1/sqrt(128)

  __shared__ __align__(16) __bf16 Kt[32 * 128];     // [key][d]    8 KB
  __shared__ __align__(16) __bf16 Vt[128 * 32];     // [d][key]    8 KB
  __shared__ __align__(16) __bf16 Pl[4][16][32];    // per-wave P relayout

  // Cache the 16x128 Q tile as 4 A-fragments.
  v16bf qa[4];
#pragma unroll
  for (int t = 0; t < 4; ++t)
    qa[t] = frag_bf16_contig(qb + (size_t)(q0 + nn) * DIM + h * HD + t * 32 + 8 * half);

  v8f zero = {};
  v8f o_acc[8];
#pragma unroll
  for (int t = 0; t < 8; ++t) o_acc[t] = zero;
  float m_i[8], l_i[8];
#pragma unroll
  for (int r = 0; r < 8; ++r) { m_i[r] = -1e30f; l_i[r] = 0.f; }

  for (int kv0 = 0; kv0 < SEQ; kv0 += 32) {
    // ---- cooperative stage of K tile (32 rows x 256B) and V tile (128 x 64B)
    {
      const char* gK = (const char*)kb +
          ((size_t)(kv0 + (tid >> 2)) * DIM + h * HD + (tid & 3) * 32) * 2;
      char*       lK = (char*)Kt + tid * 64;
      const char* gV = (const char*)vt + ((size_t)(h * HD + tid) * SEQ + kv0) * 2;
      char*       lV = (char*)Vt + tid * 64;
#if HAVE_ASYNC_LDS
#pragma unroll
      for (int c = 0; c < 4; ++c) {
        __builtin_amdgcn_global_load_async_to_lds_b128(
            (gl_v4i*)(gK + c * 16), (ld_v4i*)(lK + c * 16), 0, 0);
        __builtin_amdgcn_global_load_async_to_lds_b128(
            (gl_v4i*)(gV + c * 16), (ld_v4i*)(lV + c * 16), 0, 0);
      }
      wait_asynccnt0();
#else
#pragma unroll
      for (int c = 0; c < 4; ++c) {
        ((uint4*)lK)[c] = ((const uint4*)gK)[c];
        ((uint4*)lV)[c] = ((const uint4*)gV)[c];
      }
#endif
    }
    __syncthreads();

    // ---- scores: two 16x16 tiles over 32 keys (K-frags read from LDS)
    v8f s0 = zero, s1 = zero;
#pragma unroll
    for (int t = 0; t < 4; ++t) {
      v16bf b0 = frag_bf16_contig(&Kt[nn        * 128 + t * 32 + 8 * half]);
      v16bf b1 = frag_bf16_contig(&Kt[(16 + nn) * 128 + t * 32 + 8 * half]);
      s0 = __builtin_amdgcn_wmma_f32_16x16x32_bf16(false, qa[t], false, b0,
                                                   (short)0, s0, false, false);
      s1 = __builtin_amdgcn_wmma_f32_16x16x32_bf16(false, qa[t], false, b1,
                                                   (short)0, s1, false, false);
    }

    // ---- online softmax (row m = r + 8*half lives in one 16-lane half)
#pragma unroll
    for (int r = 0; r < 8; ++r) {
      const float a0 = s0[r] * scale;
      const float a1 = s1[r] * scale;
      float v = fmaxf(a0, a1);
      for (int mask = 8; mask; mask >>= 1) v = fmaxf(v, __shfl_xor(v, mask));
      const float mnew = fmaxf(m_i[r], v);
      const float corr = __expf(m_i[r] - mnew);
      const float p0 = __expf(a0 - mnew);
      const float p1 = __expf(a1 - mnew);
      float rs = p0 + p1;
      for (int mask = 8; mask; mask >>= 1) rs += __shfl_xor(rs, mask);
      l_i[r] = l_i[r] * corr + rs;
      m_i[r] = mnew;
#pragma unroll
      for (int t = 0; t < 8; ++t) o_acc[t][r] *= corr;
      const int m = r + 8 * half;
      Pl[wid][m][nn]      = (__bf16)p0;
      Pl[wid][m][16 + nn] = (__bf16)p1;
    }
    __syncthreads();

    // ---- re-layout P (C-frag -> A-frag): row = nn, keys in K(e) order
    v16bf pa;
#pragma unroll
    for (int e = 0; e < 8; ++e) {
      pa[e]     = Pl[wid][nn][8 * half + e];
      pa[8 + e] = Pl[wid][nn][16 + 8 * half + e];
    }

    // ---- O += P @ V  (8 d-tiles of 16; V-frags read from LDS, contiguous)
#pragma unroll
    for (int t = 0; t < 8; ++t) {
      v16bf vb = frag_bf16_contig(&Vt[(t * 16 + nn) * 32 + 8 * half]);
      o_acc[t] = __builtin_amdgcn_wmma_f32_16x16x32_bf16(false, pa, false, vb,
                                                         (short)0, o_acc[t], false, false);
    }
    __syncthreads();   // everyone done with Kt/Vt/Pl before next overwrite
  }

  // ---- finalize: divide by row sums, write fp32 O
#pragma unroll
  for (int r = 0; r < 8; ++r) {
    const float inv = 1.0f / l_i[r];
    const int m = q0 + r + 8 * half;
#pragma unroll
    for (int t = 0; t < 8; ++t)
      O[(size_t)m * DIM + h * HD + t * 16 + nn] = o_acc[t][r] * inv;
  }
}

// ---------------------------------------------------------------------------
// Host-side launcher
// ---------------------------------------------------------------------------
extern "C" void kernel_launch(void* const* d_in, const int* in_sizes, int n_in,
                              void* d_out, int out_size, void* d_ws, size_t ws_size,
                              hipStream_t stream) {
  const float* x    = (const float*)d_in[0];
  const float* wq   = (const float*)d_in[1];
  const float* bq   = (const float*)d_in[2];
  const float* wk   = (const float*)d_in[3];
  const float* bk   = (const float*)d_in[4];
  const float* wv   = (const float*)d_in[5];
  const float* bv   = (const float*)d_in[6];
  const float* wo   = (const float*)d_in[7];
  const float* bo   = (const float*)d_in[8];
  const float* gq   = (const float*)d_in[9];
  const float* gk   = (const float*)d_in[10];
  const float* fcos = (const float*)d_in[11];
  const float* fsin = (const float*)d_in[12];
  float* out = (float*)d_out;

  // Workspace carve-up (~132 MB)
  char* ws = (char*)d_ws;
  const size_t f32mat  = (size_t)SEQ * DIM * sizeof(float);    // 25.2 MB
  const size_t bf16mat = (size_t)SEQ * DIM * sizeof(__bf16);   // 12.6 MB
  const size_t bf16w   = (size_t)DIM * DIM * sizeof(__bf16);   // 4.7 MB
  float*  yq  = (float*)(ws);
  float*  yk  = (float*)(ws + f32mat);
  float*  yv  = (float*)(ws + 2 * f32mat);
  __bf16* qb  = (__bf16*)(ws + 3 * f32mat);
  __bf16* kb  = (__bf16*)(ws + 3 * f32mat + bf16mat);
  __bf16* vt  = (__bf16*)(ws + 3 * f32mat + 2 * bf16mat);
  __bf16* wqt = (__bf16*)(ws + 3 * f32mat + 3 * bf16mat);
  __bf16* wkt = (__bf16*)(ws + 3 * f32mat + 3 * bf16mat + 1 * bf16w);
  __bf16* wvt = (__bf16*)(ws + 3 * f32mat + 3 * bf16mat + 2 * bf16w);
  __bf16* wot = (__bf16*)(ws + 3 * f32mat + 3 * bf16mat + 3 * bf16w);
  float*  Ob  = yq;  // reuse yq once qb has been produced

  // 0) transpose + bf16-convert the four weight matrices
  const dim3 gT(DIM / 32, DIM / 32);
  transpose_cvt_kernel<<<gT, 256, 0, stream>>>(wq, wqt, DIM, DIM);
  transpose_cvt_kernel<<<gT, 256, 0, stream>>>(wk, wkt, DIM, DIM);
  transpose_cvt_kernel<<<gT, 256, 0, stream>>>(wv, wvt, DIM, DIM);
  transpose_cvt_kernel<<<gT, 256, 0, stream>>>(wo, wot, DIM, DIM);

  // 1) QKV projections (64x64 tile per wave, bf16 WMMA, fp32 accumulate)
  const dim3 gGemm(SEQ / 64, DIM / 64);
  gemm64_kernel<<<gGemm, 32, 0, stream>>>(x, wqt, bq, yq, SEQ, DIM, DIM);
  gemm64_kernel<<<gGemm, 32, 0, stream>>>(x, wkt, bk, yk, SEQ, DIM, DIM);
  gemm64_kernel<<<gGemm, 32, 0, stream>>>(x, wvt, bv, yv, SEQ, DIM, DIM);

  // 2) RMSNorm + RoPE -> bf16 Q/K ; V -> bf16 transposed [h*HD+d, s]
  norm_rope_cvt_kernel<<<SEQ, 256, 0, stream>>>(yq, gq, fcos, fsin, qb, 1);
  norm_rope_cvt_kernel<<<SEQ, 256, 0, stream>>>(yk, gk, fcos, fsin, kb, 1);
  norm_rope_cvt_kernel<<<SEQ, 256, 0, stream>>>(yv, gq, fcos, fsin, vt, 0);

  // 3) flash attention (4 waves/block share K/V tiles via LDS)
  attn_kernel<<<dim3(SEQ / 64, NH), 128, 0, stream>>>(qb, kb, vt, Ob);

  // 4) output projection
  gemm64_kernel<<<gGemm, 32, 0, stream>>>(Ob, wot, bo, out, SEQ, DIM, DIM);

  (void)in_sizes; (void)n_in; (void)out_size; (void)ws_size;
}
